// SeqCondAttention_89558658056581
// MI455X (gfx1250) — matrix operand order repaired and checked by reading, not compiled, not run on hardware
//
#include <hip/hip_runtime.h>
#include <hip/hip_bf16.h>

typedef __bf16 bf16;
typedef __attribute__((ext_vector_type(16))) __bf16 v16bf;
typedef __attribute__((ext_vector_type(8)))  __bf16 v8bf;
typedef __attribute__((ext_vector_type(8)))  float  v8f;

#define DEVINL __device__ __forceinline__

// ---------------- problem constants ----------------
namespace cfg {
constexpr int D_MODEL = 1536, K = 12, K_Q = 6, H = 128, KC = 4;
constexpr int DIM_MEM = 1536, DIM_QT = 1536;
constexpr int DIM_SKIP = 384, DIM_MEM_TOT = 1548, DIM_CONV = 3084, DIM_TOT = 3480;
constexpr int DIM_SW = 9216;              // K * 768
constexpr int Bb = 4, Ll = 2048, BL = Bb * Ll;   // 8192 rows
constexpr int OUTF = 3072;                // K * 2 * H
constexpr int YW = 4608;                  // K * 384
constexpr int NCH = 16, CHL = 128;        // scan chunking: NCH*CHL == Ll
}
using namespace cfg;

// ---------------- CDNA5 async global->LDS helpers ----------------
DEVINL unsigned lds_off_u32(const void* p) {
  // flat address of a __shared__ object: low 32 bits are the LDS byte offset
  return (unsigned)(unsigned long long)p;
}
DEVINL void async_b128(unsigned ldsoff, const void* g) {
  asm volatile("global_load_async_to_lds_b128 %0, %1, off"
               :: "v"(ldsoff), "v"(g) : "memory");
}
DEVINL void wait_async0() {
  asm volatile("s_wait_asynccnt 0" ::: "memory");
}
// async loads complete in order: waiting <=4 guarantees all but the newest
// staged tile (4 copies/thread) are resident.
DEVINL void wait_async4() {
  asm volatile("s_wait_asynccnt 4" ::: "memory");
}

// ---------------- WMMA bf16 NT GEMM ----------------
// C(M x N) = A(M x K) * B(N x K)^T ; A,B bf16 row-major, C f32.
// Triple-buffered LDS, async global->LDS staging (2 tiles in flight),
// 8 back-to-back WMMAs per wave per K-step, one barrier per K-step.
// B over-reads past N are tolerated (workspace-backed); C store is N-guarded.
constexpr int BM = 128, BN = 128, BKt = 32, LPAD = 8, LSTR = BKt + LPAD;

__global__ __launch_bounds__(256) void gemm_bf16_nt(
    const bf16* __restrict__ A, long lda, long sA,
    const bf16* __restrict__ Bm, long ldb, long sB,
    float* __restrict__ C, long ldc, long sC,
    int M, int N, int Kd)
{
  __shared__ __align__(16) bf16 As[3][BM * LSTR];
  __shared__ __align__(16) bf16 Bs[3][BN * LSTR];
  A  += (long)blockIdx.z * sA;
  Bm += (long)blockIdx.z * sB;
  C  += (long)blockIdx.z * sC;
  const int n0 = blockIdx.x * BN;
  const int m0 = blockIdx.y * BM;
  const int tid  = threadIdx.x;
  const int lane = tid & 31;
  const int wave = tid >> 5;
  const int wm = (wave & 1) * 64;     // 2x4 wave grid: 64x32 per wave
  const int wn = (wave >> 1) * 32;
  const int lr = lane & 15;
  const int lh = lane >> 4;

  // staging map: unit u in {tid, tid+256}: row = u>>2 (0..127), col = (u&3)*8
  const int r0 = tid >> 2, c0 = (tid & 3) * 8;

  v8f acc[4][2];
  for (int i = 0; i < 4; i++)
    for (int j = 0; j < 2; j++)
      for (int e = 0; e < 8; e++) acc[i][j][e] = 0.f;

  auto stage = [&](int buf, int k0) {
    async_b128(lds_off_u32(&As[buf][r0 * LSTR + c0]),
               A + (long)(m0 + r0) * lda + k0 + c0);
    async_b128(lds_off_u32(&As[buf][(r0 + 64) * LSTR + c0]),
               A + (long)(m0 + r0 + 64) * lda + k0 + c0);
    async_b128(lds_off_u32(&Bs[buf][r0 * LSTR + c0]),
               Bm + (long)(n0 + r0) * ldb + k0 + c0);
    async_b128(lds_off_u32(&Bs[buf][(r0 + 64) * LSTR + c0]),
               Bm + (long)(n0 + r0 + 64) * ldb + k0 + c0);
  };

  const int nk = Kd / BKt;
  stage(0, 0);
  if (nk > 1) stage(1, BKt);

  for (int kt = 0; kt < nk; kt++) {
    const int buf = kt % 3;
    // tile kt resident when at most the newest staged tile is outstanding
    if (kt + 1 < nk) wait_async4(); else wait_async0();
    // publishes tile kt's LDS writes; also fences last step's fragment reads
    __syncthreads();
    if (kt + 2 < nk) stage((kt + 2) % 3, (kt + 2) * BKt);

    // Fragment layouts (wave32, 16-bit 16x32 A / 32x16 B, per ISA 7.12.2):
    //  A lane: row = lane&15; elems 0..7 -> k = lh*8+0..7, 8..15 -> k = 16+lh*8+0..7
    //  B lane: col = lane&15; elems 0..7 -> k = lh*8+0..7, 8..15 -> k = 16+lh*8+0..7
    v16bf bq[2], af[4];
    for (int j = 0; j < 2; j++) {
      const bf16* rp = &Bs[buf][(wn + j * 16 + lr) * LSTR];
      v8bf lo = *(const v8bf*)(rp + lh * 8);
      v8bf hi = *(const v8bf*)(rp + 16 + lh * 8);
      for (int e = 0; e < 8; e++) { bq[j][e] = lo[e]; bq[j][8 + e] = hi[e]; }
    }
    for (int i = 0; i < 4; i++) {
      const bf16* rp = &As[buf][(wm + i * 16 + lr) * LSTR];
      v8bf lo = *(const v8bf*)(rp + lh * 8);
      v8bf hi = *(const v8bf*)(rp + 16 + lh * 8);
      for (int e = 0; e < 8; e++) { af[i][e] = lo[e]; af[i][8 + e] = hi[e]; }
    }
    // 8 back-to-back WMMAs, disjoint accumulators (no D->A/B hazards)
    for (int i = 0; i < 4; i++)
      for (int j = 0; j < 2; j++)
        acc[i][j] = __builtin_amdgcn_wmma_f32_16x16x32_bf16(
            false, af[i], false, bq[j], (short)0, acc[i][j], false, false);
  }

  // C/D layout: vgpr v, lanes 0-15 -> M=v, N=lane; lanes 16-31 -> M=8+v
  for (int i = 0; i < 4; i++) {
    const int rbase = m0 + wm + i * 16 + lh * 8;
    for (int j = 0; j < 2; j++) {
      const int n = n0 + wn + j * 16 + lr;
      if (n < N) {
        for (int v = 0; v < 8; v++)
          C[(long)(rbase + v) * ldc + n] = acc[i][j][v];
      }
    }
  }
}

// ---------------- conversions ----------------
__global__ void cvt_bf16_kernel(const float* __restrict__ in, bf16* __restrict__ out, long n) {
  long i = (long)blockIdx.x * blockDim.x + threadIdx.x;
  long stride = (long)gridDim.x * blockDim.x;
  for (; i < n; i += stride) out[i] = (bf16)in[i];
}

__global__ void cskip_cvt_kernel(const float* __restrict__ z_all, bf16* __restrict__ out, long n) {
  long i = (long)blockIdx.x * blockDim.x + threadIdx.x;
  long stride = (long)gridDim.x * blockDim.x;
  for (; i < n; i += stride) {
    long row = i / DIM_SKIP;
    int c = (int)(i % DIM_SKIP);
    out[i] = (bf16)z_all[row * DIM_TOT + DIM_CONV + c];
  }
}

// W_readout (K,256,768) f32 -> (K,768,256) bf16 (transposed for NT GEMM)
__global__ void wro_t_kernel(const float* __restrict__ w, bf16* __restrict__ out) {
  long n = (long)K * 768 * 256;
  long i = (long)blockIdx.x * blockDim.x + threadIdx.x;
  long stride = (long)gridDim.x * blockDim.x;
  for (; i < n; i += stride) {
    long r = i % (768 * 256);
    long k = i / (768 * 256);
    int nn = (int)(r / 256), c = (int)(r % 256);
    out[i] = (bf16)w[(k * 256 + c) * 768 + nn];
  }
}

// ---------------- depthwise causal conv (KC=4) + SiLU ----------------
__global__ void conv_silu_kernel(const float* __restrict__ z_all, const float* __restrict__ cw,
                                 float* __restrict__ z) {
  int c = blockIdx.x * blockDim.x + threadIdx.x;
  int row = blockIdx.y;                 // b*L + t
  if (c >= DIM_CONV) return;
  int t = row & (Ll - 1);
  float w0 = cw[c * KC + 0], w1 = cw[c * KC + 1], w2 = cw[c * KC + 2], w3 = cw[c * KC + 3];
  const float* base = z_all + (long)row * DIM_TOT + c;
  float a = base[0] * w3;
  if (t >= 1) a += base[-(long)DIM_TOT] * w2;
  if (t >= 2) a += base[-2L * DIM_TOT]  * w1;
  if (t >= 3) a += base[-3L * DIM_TOT]  * w0;
  z[(long)row * DIM_CONV + c] = a / (1.f + __expf(-a));   // silu
}

// ---------------- per-(b,t,k) rms scale + p_w ----------------
__global__ __launch_bounds__(128) void rqk_pw_kernel(
    const float* __restrict__ z, const float* __restrict__ decay,
    const float* __restrict__ ssc, const float* __restrict__ sbi,
    float* __restrict__ rqk, float* __restrict__ pw) {
  int row = blockIdx.x;
  int k = blockIdx.y;
  int h = threadIdx.x;
  float v = z[(long)row * DIM_CONV + k * H + h];
  float s = v * v;
  for (int o = 16; o > 0; o >>= 1) s += __shfl_xor(s, o, 32);
  __shared__ float p[4];
  if ((h & 31) == 0) p[h >> 5] = s;
  __syncthreads();
  if (h == 0) {
    float tot = p[0] + p[1] + p[2] + p[3];
    rqk[(long)row * K + k] = rsqrtf(tot * (1.f / H) + 1e-6f);
    int t = row & (Ll - 1);
    float sraw  = z[(long)row * DIM_CONV + DIM_MEM + k];
    float score = ssc[k] * sraw + sbi[k];
    float r     = fmaxf(score, 0.f);
    float slope = log1pf(__expf(decay[k]));                 // softplus
    float pv    = r * r * __expf(-slope * (float)(Ll - 1 - t));
    pw[(long)row * K + k] = fminf(fmaxf(pv, 1e-6f), 1000.f);
  }
}

// per-(b,t) rms scale for q (1536 wide)
__global__ __launch_bounds__(256) void rq_kernel(const float* __restrict__ z, float* __restrict__ rq) {
  int row = blockIdx.x;
  int tid = threadIdx.x;
  const float* q = z + (long)row * DIM_CONV + DIM_MEM_TOT;
  float s = 0.f;
  for (int i = 0; i < 6; i++) { float v = q[tid + i * 256]; s += v * v; }
  for (int o = 16; o > 0; o >>= 1) s += __shfl_xor(s, o, 32);
  __shared__ float p[8];
  if ((tid & 31) == 0) p[tid >> 5] = s;
  __syncthreads();
  if (tid == 0) {
    float tot = 0;
    for (int i = 0; i < 8; i++) tot += p[i];
    rq[row] = rsqrtf(tot * (1.f / DIM_QT) + 1e-6f);
  }
}

// ---------------- chunked prefix scan, pass 1: chunk totals ----------------
__global__ __launch_bounds__(128) void scan_pass1_kernel(
    const float* __restrict__ z, const float* __restrict__ rqk, const float* __restrict__ pw,
    const float* __restrict__ knw, const float* __restrict__ theta_raw,
    const float* __restrict__ phase_scale, float* __restrict__ totals) {
  int c  = blockIdx.x % NCH;
  int bk = blockIdx.x / NCH;          // b*K + k
  int k  = bk % K, b = bk / K;
  int h  = threadIdx.x;
  float th = 0.001f + 2.999f / (1.f + __expf(-theta_raw[k * H + h]));
  float ps = phase_scale[k];
  float kw = knw[h];
  float re = 0.f, im = 0.f;
  int t0 = c * CHL;
  for (int tt = 0; tt < CHL; tt++) {
    long row = (long)b * Ll + (t0 + tt);
    float kv  = z[row * DIM_CONV + k * H + h] * rqk[row * K + k] * kw;
    float ks  = kv * ps;
    float phi = ks / (1.f + fabsf(ks)) * th;
    float kvw = kv * pw[row * K + k];
    float sn, cs;
    __sincosf(phi, &sn, &cs);
    re += kvw * cs;
    im += kvw * sn;
  }
  long tb = ((long)bk * NCH + c) * (2 * H);
  totals[tb + h]     = re;
  totals[tb + H + h] = im;
}

// ---------------- scan pass 2: carry-in + fused q readout ----------------
__global__ __launch_bounds__(128) void scan_pass2_kernel(
    const float* __restrict__ z, const float* __restrict__ rqk, const float* __restrict__ pw,
    const float* __restrict__ rq, const float* __restrict__ knw, const float* __restrict__ qnw,
    const float* __restrict__ theta_raw, const float* __restrict__ w_int_raw,
    const float* __restrict__ phase_scale, const float* __restrict__ totals,
    float* __restrict__ out_flat) {
  int c  = blockIdx.x % NCH;
  int bk = blockIdx.x / NCH;
  int k  = bk % K, b = bk / K;
  int h  = threadIdx.x;
  float th = 0.001f + 2.999f / (1.f + __expf(-theta_raw[k * H + h]));
  float ps = phase_scale[k];
  float kw = knw[h];
  float wi = __expf(fminf(fmaxf(w_int_raw[k * H + h], -5.f), 5.f));
  int kq = k >> 1;
  int qoff = DIM_MEM_TOT + kq * (2 * H);
  float qw_re = qnw[kq * (2 * H) + 2 * h];
  float qw_im = qnw[kq * (2 * H) + 2 * h + 1];
  float re = 0.f, im = 0.f;
  const float* tb = totals + (long)bk * NCH * (2 * H);
  for (int c2 = 0; c2 < c; c2++) { re += tb[c2 * (2 * H) + h]; im += tb[c2 * (2 * H) + H + h]; }
  int t0 = c * CHL;
  for (int tt = 0; tt < CHL; tt++) {
    long row = (long)b * Ll + (t0 + tt);
    const float* zr = z + row * DIM_CONV;
    float kv  = zr[k * H + h] * rqk[row * K + k] * kw;
    float ks  = kv * ps;
    float phi = ks / (1.f + fabsf(ks)) * th;
    float kvw = kv * pw[row * K + k];
    float sn, cs;
    __sincosf(phi, &sn, &cs);
    re += kvw * cs;
    im += kvw * sn;
    float rv  = rq[row];
    float qre = zr[qoff + 2 * h]     * rv * qw_re;
    float qim = zr[qoff + 2 * h + 1] * rv * qw_im;
    float* o = out_flat + row * OUTF + k * (2 * H);
    o[h]     = (re * qre + im * qim) * wi;
    o[H + h] = (im * qre - re * qim) * wi;
  }
}

// ---------------- gate GEMM (inner dim 12) + gated rmsnorm -> bf16 oc ----------------
__global__ __launch_bounds__(256) void gated_norm_kernel(
    const float* __restrict__ z_all, const float* __restrict__ out_flat,
    const float* __restrict__ W_gate, const float* __restrict__ gnw,
    bf16* __restrict__ oc) {
  int row = blockIdx.x;
  int tid = threadIdx.x;
  __shared__ float g[12];
  __shared__ float red[9];
  if (tid < 12) g[tid] = z_all[(long)row * DIM_TOT + (DIM_CONV + DIM_SKIP) + tid];
  __syncthreads();
  float xg[12];
  float ss = 0.f;
  for (int i = 0; i < 12; i++) {
    int j = tid + i * 256;
    float gn = 0.f;
    for (int q = 0; q < 12; q++) gn += g[q] * W_gate[(long)j * 12 + q];
    float x = out_flat[(long)row * OUTF + j] * gn / (1.f + __expf(-gn));  // * silu(gate)
    xg[i] = x;
    ss += x * x;
  }
  for (int o = 16; o > 0; o >>= 1) ss += __shfl_xor(ss, o, 32);
  if ((tid & 31) == 0) red[tid >> 5] = ss;
  __syncthreads();
  if (tid == 0) {
    float t = 0;
    for (int i = 0; i < 8; i++) t += red[i];
    red[8] = rsqrtf(t * (1.f / OUTF) + 1e-6f);
  }
  __syncthreads();
  float r = red[8];
  for (int i = 0; i < 12; i++) {
    int j = tid + i * 256;
    oc[(long)row * OUTF + j] = (bf16)(xg[i] * r * gnw[j]);
  }
}

// ---------------- highway combine -> bf16 y ----------------
__global__ void combine_kernel(const float* __restrict__ y_spec, const float* __restrict__ y_skip,
                               const float* __restrict__ hs, bf16* __restrict__ y) {
  long n = (long)BL * YW;
  long i = (long)blockIdx.x * blockDim.x + threadIdx.x;
  long stride = (long)gridDim.x * blockDim.x;
  for (; i < n; i += stride) {
    long row = i / YW;
    int r2 = (int)(i % YW);
    int k = r2 / 384, j = r2 % 384;
    long base = row * (long)DIM_SW + (long)k * 768 + j;
    float sv = y_spec[base], sg = y_spec[base + 384];
    float kv = y_skip[base], kg = y_skip[base + 384];
    float hh  = hs[k];
    float val = sv + kv * hh;
    float gt  = sg + kg * hh;
    y[i] = (bf16)(val / (1.f + __expf(-gt)));   // val * sigmoid(gate)
  }
}

// ---------------- host ----------------
extern "C" void kernel_launch(void* const* d_in, const int* in_sizes, int n_in,
                              void* d_out, int out_size, void* d_ws, size_t ws_size,
                              hipStream_t stream) {
  (void)in_sizes; (void)n_in; (void)out_size; (void)ws_size;
  const float* x         = (const float*)d_in[0];
  const float* W_in      = (const float*)d_in[1];
  const float* conv_w    = (const float*)d_in[2];
  const float* knw       = (const float*)d_in[3];
  const float* qnw       = (const float*)d_in[4];
  const float* theta_raw = (const float*)d_in[5];
  const float* w_int_raw = (const float*)d_in[6];
  const float* decay     = (const float*)d_in[7];
  const float* ssc       = (const float*)d_in[8];
  const float* sbi       = (const float*)d_in[9];
  const float* ph        = (const float*)d_in[10];
  const float* W_ro      = (const float*)d_in[11];
  const float* W_gate    = (const float*)d_in[12];
  const float* gnw       = (const float*)d_in[13];
  const float* W_skip    = (const float*)d_in[14];
  const float* hsc       = (const float*)d_in[15];
  const float* W_out     = (const float*)d_in[16];
  float* out = (float*)d_out;

  char* ws = (char*)d_ws;
  size_t off = 0;
  auto alloc = [&](size_t bytes) -> char* {
    char* p = ws + off;
    off = (off + bytes + 255) & ~(size_t)255;
    return p;
  };
  bf16*  W_in_bf   = (bf16*) alloc((size_t)DIM_TOT * D_MODEL * 2);
  bf16*  W_skip_bf = (bf16*) alloc((size_t)DIM_SW * DIM_SKIP * 2);
  bf16*  W_out_bf  = (bf16*) alloc((size_t)D_MODEL * YW * 2);
  bf16*  W_ro_t    = (bf16*) alloc((size_t)K * 768 * 256 * 2);
  bf16*  x_bf      = (bf16*) alloc((size_t)BL * D_MODEL * 2);
  float* rq        = (float*)alloc((size_t)BL * 4);
  float* rqk       = (float*)alloc((size_t)BL * K * 4);
  float* pw        = (float*)alloc((size_t)BL * K * 4);
  float* totals    = (float*)alloc((size_t)Bb * K * NCH * 2 * H * 4);
  bf16*  oc_bf     = (bf16*) alloc((size_t)BL * OUTF * 2);
  bf16*  cskip_bf  = (bf16*) alloc((size_t)BL * DIM_SKIP * 2);
  float* y_spec    = (float*)alloc((size_t)BL * DIM_SW * 4);
  bf16*  y_bf      = (bf16*) alloc((size_t)BL * YW * 2);
  // big region: z_all | z | out_flat, later reused for y_skip (all dead by then)
  size_t szA = ((size_t)BL * DIM_TOT * 4 + 255) & ~(size_t)255;
  size_t szZ = ((size_t)BL * DIM_CONV * 4 + 255) & ~(size_t)255;
  size_t szO = ((size_t)BL * OUTF * 4 + 255) & ~(size_t)255;
  char* big = alloc(szA + szZ + szO);
  float* z_all    = (float*)big;
  float* zbuf     = (float*)(big + szA);
  float* out_flat = (float*)(big + szA + szZ);
  float* y_skip   = (float*)big;

  // precision conversions (weights + x), done per call (deterministic)
  cvt_bf16_kernel<<<1024, 256, 0, stream>>>(x, x_bf, (long)BL * D_MODEL);
  cvt_bf16_kernel<<<1024, 256, 0, stream>>>(W_in, W_in_bf, (long)DIM_TOT * D_MODEL);
  cvt_bf16_kernel<<<1024, 256, 0, stream>>>(W_skip, W_skip_bf, (long)DIM_SW * DIM_SKIP);
  cvt_bf16_kernel<<<1024, 256, 0, stream>>>(W_out, W_out_bf, (long)D_MODEL * YW);
  wro_t_kernel<<<1024, 256, 0, stream>>>(W_ro, W_ro_t);

  // (1) z_all = x @ W_in^T    8192 x 3480, K=1536
  gemm_bf16_nt<<<dim3((DIM_TOT + BN - 1) / BN, BL / BM, 1), 256, 0, stream>>>(
      x_bf, (long)D_MODEL, 0L, W_in_bf, (long)D_MODEL, 0L, z_all, (long)DIM_TOT, 0L,
      BL, DIM_TOT, D_MODEL);

  // (2) depthwise causal conv + silu
  conv_silu_kernel<<<dim3((DIM_CONV + 255) / 256, BL, 1), 256, 0, stream>>>(z_all, conv_w, zbuf);

  // c_skip slice -> bf16
  cskip_cvt_kernel<<<1024, 256, 0, stream>>>(z_all, cskip_bf, (long)BL * DIM_SKIP);

  // (3) norms + p_w precompute
  rqk_pw_kernel<<<dim3(BL, K, 1), 128, 0, stream>>>(zbuf, decay, ssc, sbi, rqk, pw);
  rq_kernel<<<BL, 256, 0, stream>>>(zbuf, rq);

  // (4)+(5) chunked prefix scan with fused readout -> out_flat
  scan_pass1_kernel<<<Bb * K * NCH, 128, 0, stream>>>(zbuf, rqk, pw, knw, theta_raw, ph, totals);
  scan_pass2_kernel<<<Bb * K * NCH, 128, 0, stream>>>(zbuf, rqk, pw, rq, knw, qnw,
                                                      theta_raw, w_int_raw, ph, totals, out_flat);

  // (6) gate GEMM (inner 12) + gated rmsnorm -> oc bf16
  gated_norm_kernel<<<BL, 256, 0, stream>>>(z_all, out_flat, W_gate, gnw, oc_bf);

  // (7) y_spec: batched per-k GEMM  (8192 x 768, K=256) x 12
  gemm_bf16_nt<<<dim3(768 / BN, BL / BM, K), 256, 0, stream>>>(
      oc_bf, (long)OUTF, (long)(2 * H), W_ro_t, 256L, (long)768 * 256,
      y_spec, (long)DIM_SW, 768L, BL, 768, 2 * H);

  // (8) y_skip GEMM (aliases z_all region; z_all/zbuf/out_flat dead now)
  gemm_bf16_nt<<<dim3(DIM_SW / BN, BL / BM, 1), 256, 0, stream>>>(
      cskip_bf, (long)DIM_SKIP, 0L, W_skip_bf, (long)DIM_SKIP, 0L,
      y_skip, (long)DIM_SW, 0L, BL, DIM_SW, DIM_SKIP);

  // highway + sigmoid -> y bf16
  combine_kernel<<<2048, 256, 0, stream>>>(y_spec, y_skip, hsc, y_bf);

  // (9) final GEMM -> d_out (f32)   8192 x 1536, K=4608
  gemm_bf16_nt<<<dim3(D_MODEL / BN, BL / BM, 1), 256, 0, stream>>>(
      y_bf, (long)YW, 0L, W_out_bf, (long)YW, 0L, out, (long)D_MODEL, 0L,
      BL, D_MODEL, YW);
}
